// TinyGPT10k_26276609917394
// MI455X (gfx1250) — matrix operand order repaired and checked by reading, not compiled, not run on hardware
//
#include <hip/hip_runtime.h>
#include <math.h>

typedef __attribute__((ext_vector_type(16))) _Float16 v16h;
typedef __attribute__((ext_vector_type(8)))  float    v8f;

#define WPB   4      // waves (sequences) per block
#define SEQT  32     // tokens per sequence
#define DIM   16     // model dim
#define NVOC  96     // vocab
#define NTILE 13     // staged weight B-tiles

// Weight tile ids in LDS staging area
// 0-2: in_w (qkv, N=48)  3: out_w  4-5: w1 (N=32)  6: w2 (K=32)  7-12: tok_emb (N=96)

// ---- WMMA operand loaders (contiguous k order: k = 16*half + i, wave32) ----------

// A tile 16x32 f16 from f32 LDS, K=16 zero-padded.
__device__ __forceinline__ v16h load_A_k16(const float* buf, int stride,
                                           int mt, int lrow, int half) {
  const float* p = buf + (mt * 16 + lrow) * stride;
  v16h a;
#pragma unroll
  for (int i = 0; i < 16; ++i) a[i] = (half == 0) ? (_Float16)p[i] : (_Float16)0.0f;
  return a;
}

// A tile 16x32 f16 from f32 LDS, full K=32.
__device__ __forceinline__ v16h load_A_k32(const float* buf, int stride,
                                           int mt, int lrow, int half) {
  const float* p = buf + (mt * 16 + lrow) * stride + half * 16;
  v16h a;
#pragma unroll
  for (int i = 0; i < 16; ++i) a[i] = (_Float16)p[i];
  return a;
}

// Pre-staged f16 B tile: per-lane 16-half image, 2x ds_load_b128.
__device__ __forceinline__ v16h load_B_w(const _Float16* wlds, int tile, int lane) {
  return *(const v16h*)(wlds + (tile * 32 + lane) * 16);
}

// B tile from f32 LDS rows (X @ Wt with W rows contiguous), K=16 padded.
// B[k][n] = src[(n0+n)*stride + coff + k]
__device__ __forceinline__ v16h load_B_lds_k16(const float* src, int stride,
                                               int n0, int coff,
                                               int lrow, int half) {
  v16h b;
  if (half == 0) {
    const float* p = src + (n0 + lrow) * stride + coff;
#pragma unroll
    for (int i = 0; i < 16; ++i) b[i] = (_Float16)p[i];
  } else {
#pragma unroll
    for (int i = 0; i < 16; ++i) b[i] = (_Float16)0.0f;
  }
  return b;
}

// B tile for probs@v: B[k][n] = qkv[k*48 + 32 + n], K=32 (column-strided LDS reads)
__device__ __forceinline__ v16h load_B_vT(const float* qkv, int lrow, int half) {
  const int kb = half * 16;
  v16h b;
#pragma unroll
  for (int i = 0; i < 16; ++i)
    b[i] = (_Float16)qkv[(kb + i) * 48 + 32 + lrow];
  return b;
}

__device__ __forceinline__ v8f splat8(float v) {
  v8f c;
#pragma unroll
  for (int r = 0; r < 8; ++r) c[r] = v;
  return c;
}

// ---------------------------------------------------------------------------

struct __align__(32) SMem {
  _Float16 wlds[NTILE * 32 * 16];   // 13 KB: shared f16 weight B-tiles
  float    act[WPB][2560];          // per-wave: qkv[32][48] | sbuf[32][32]
};

__global__ __launch_bounds__(WPB * 32) void tinygpt_fused_kernel(
    const int* __restrict__ idx,
    const float* __restrict__ tok_emb, const float* __restrict__ pos_emb,
    const float* __restrict__ ln1_g, const float* __restrict__ ln1_b,
    const float* __restrict__ in_w,  const float* __restrict__ in_b,
    const float* __restrict__ out_w, const float* __restrict__ out_b,
    const float* __restrict__ ln2_g, const float* __restrict__ ln2_b,
    const float* __restrict__ w1, const float* __restrict__ b1,
    const float* __restrict__ w2, const float* __restrict__ b2,
    float* __restrict__ out, int nseq)
{
  __shared__ SMem sm;

  const int wave = threadIdx.x >> 5;
  const int lane = threadIdx.x & 31;
  const int half = lane >> 4;       // 0: lanes 0-15, 1: lanes 16-31
  const int lrow = lane & 15;
  int seq = blockIdx.x * WPB + wave;
  if (seq >= nseq) seq = nseq - 1;  // duplicate (identical) work, no barrier skew

  float* bufQKV = sm.act[wave];         // stride 48 (qkv) / 32 (h)
  float* sbuf   = bufQKV + 1536;        // [32][32]: x -> scores/probs -> ctx|x2
  float* bufX   = sbuf;                 // [32][16]: x -> ctx -> final x
  float* bufX2  = sbuf + 512;           // [32][16]: post-attn / post-LN2

  // ---- 0. stage all weight B-tiles as f16 (block-cooperative, once) -------
  for (int e = threadIdx.x; e < NTILE * 32; e += WPB * 32) {
    const int tile = e >> 5, ln = e & 31;
    const int hf = ln >> 4, lr = ln & 15;
    const float* W; int n0, kc;
    switch (tile) {
      case 0:  W = in_w;    n0 = 0;  kc = 16; break;
      case 1:  W = in_w;    n0 = 16; kc = 16; break;
      case 2:  W = in_w;    n0 = 32; kc = 16; break;
      case 3:  W = out_w;   n0 = 0;  kc = 16; break;
      case 4:  W = w1;      n0 = 0;  kc = 16; break;
      case 5:  W = w1;      n0 = 16; kc = 16; break;
      case 6:  W = w2;      n0 = 0;  kc = 32; break;
      default: W = tok_emb; n0 = (tile - 7) * 16; kc = 16; break;
    }
    _Float16* dst = &sm.wlds[e * 16];
    if (kc == 32) {
      const float* p = W + (n0 + lr) * 32 + hf * 16;
#pragma unroll
      for (int i = 0; i < 16; ++i) dst[i] = (_Float16)p[i];
    } else if (hf == 0) {
      const float* p = W + (n0 + lr) * 16;
#pragma unroll
      for (int i = 0; i < 16; ++i) dst[i] = (_Float16)p[i];
    } else {
#pragma unroll
      for (int i = 0; i < 16; ++i) dst[i] = (_Float16)0.0f;
    }
  }

  // ---- 1. embedding + LN1 (lane t owns token row t, fully in-lane) --------
  {
    const int t = lane;
    const int tok = idx[seq * SEQT + t];
    float xr[DIM];
#pragma unroll
    for (int d = 0; d < DIM; ++d)
      xr[d] = tok_emb[tok * DIM + d] + pos_emb[t * DIM + d];
    float m = 0.f;
#pragma unroll
    for (int d = 0; d < DIM; ++d) m += xr[d];
    m *= (1.0f / DIM);
    float var = 0.f;
#pragma unroll
    for (int d = 0; d < DIM; ++d) { float c = xr[d] - m; var += c * c; }
    var *= (1.0f / DIM);
    float rs = rsqrtf(var + 1e-5f);
#pragma unroll
    for (int d = 0; d < DIM; ++d)
      bufX[t * DIM + d] = (xr[d] - m) * rs * ln1_g[d] + ln1_b[d];
  }
  __syncthreads();

  // ---- 2. qkv = x @ in_w^T + in_b : 6 WMMAs -------------------------------
#pragma unroll
  for (int mt = 0; mt < 2; ++mt) {
    v16h a = load_A_k16(bufX, DIM, mt, lrow, half);
#pragma unroll
    for (int nt = 0; nt < 3; ++nt) {
      v16h b = load_B_w(sm.wlds, nt, lane);
      v8f  c = splat8(in_b[nt * 16 + lrow]);
      c = __builtin_amdgcn_wmma_f32_16x16x32_f16(false, a, false, b,
                                                 (short)0, c, false, false);
#pragma unroll
      for (int r = 0; r < 8; ++r)
        bufQKV[(mt * 16 + half * 8 + r) * 48 + nt * 16 + lrow] = c[r];
    }
  }
  __syncthreads();

  // ---- 3a. scores = q @ k^T : [32,16]x[16,32], 4 WMMAs (mask applied later)
#pragma unroll
  for (int mt = 0; mt < 2; ++mt) {
    v16h a = load_A_k16(bufQKV, 48, mt, lrow, half);        // q rows
#pragma unroll
    for (int nt = 0; nt < 2; ++nt) {
      v16h b = load_B_lds_k16(bufQKV, 48, nt * 16, 16, lrow, half);  // k rows
      v8f  c = splat8(0.0f);
      c = __builtin_amdgcn_wmma_f32_16x16x32_f16(false, a, false, b,
                                                 (short)0, c, false, false);
#pragma unroll
      for (int r = 0; r < 8; ++r)
        sbuf[(mt * 16 + half * 8 + r) * 32 + nt * 16 + lrow] = c[r];
    }
  }
  __syncthreads();

  // ---- 3b. causal softmax, normalized in place (lane t owns row t) --------
  {
    const int t = lane;
    float* row = sbuf + t * 32;
    float raw[SEQT];
#pragma unroll
    for (int s2 = 0; s2 < SEQT; ++s2) raw[s2] = row[s2] * 0.25f; // 1/sqrt(16)
    float m = -1e30f;
#pragma unroll
    for (int s2 = 0; s2 < SEQT; ++s2) if (s2 <= t) m = fmaxf(m, raw[s2]);
    float sum = 0.f;
#pragma unroll
    for (int s2 = 0; s2 < SEQT; ++s2) {
      float p = (s2 <= t) ? __expf(raw[s2] - m) : 0.0f;
      raw[s2] = p;
      sum += p;
    }
    float inv = 1.0f / sum;
#pragma unroll
    for (int s2 = 0; s2 < SEQT; ++s2) row[s2] = raw[s2] * inv;
  }
  __syncthreads();

  // ---- 3c. ctx = probs @ v : [32,32]x[32,16] K=32, 2 WMMAs ----------------
  {
    v16h a0 = load_A_k32(sbuf, 32, 0, lrow, half);
    v16h a1 = load_A_k32(sbuf, 32, 1, lrow, half);
    v16h bv = load_B_vT(bufQKV, lrow, half);
    v8f  c0 = splat8(0.0f), c1 = splat8(0.0f);
    c0 = __builtin_amdgcn_wmma_f32_16x16x32_f16(false, a0, false, bv,
                                                (short)0, c0, false, false);
    c1 = __builtin_amdgcn_wmma_f32_16x16x32_f16(false, a1, false, bv,
                                                (short)0, c1, false, false);
    __syncthreads();   // all probs reads done before ctx overwrites sbuf
#pragma unroll
    for (int r = 0; r < 8; ++r) {
      bufX[(half * 8 + r) * DIM + lrow]        = c0[r];
      bufX[(16 + half * 8 + r) * DIM + lrow]   = c1[r];
    }
  }
  __syncthreads();

  // ---- 4. out_proj: ctx @ out_w^T + out_b : 2 WMMAs -----------------------
#pragma unroll
  for (int mt = 0; mt < 2; ++mt) {
    v16h a = load_A_k16(bufX, DIM, mt, lrow, half);
    v16h b = load_B_w(sm.wlds, 3, lane);
    v8f  c = splat8(out_b[lrow]);
    c = __builtin_amdgcn_wmma_f32_16x16x32_f16(false, a, false, b,
                                               (short)0, c, false, false);
#pragma unroll
    for (int r = 0; r < 8; ++r)
      bufX2[(mt * 16 + half * 8 + r) * DIM + lrow] = c[r];
  }
  __syncthreads();

  // ---- 5. LN2 (in-lane per row) -------------------------------------------
  {
    const int t = lane;
    float xr[DIM];
#pragma unroll
    for (int d = 0; d < DIM; ++d) xr[d] = bufX2[t * DIM + d];
    float m = 0.f;
#pragma unroll
    for (int d = 0; d < DIM; ++d) m += xr[d];
    m *= (1.0f / DIM);
    float var = 0.f;
#pragma unroll
    for (int d = 0; d < DIM; ++d) { float c = xr[d] - m; var += c * c; }
    var *= (1.0f / DIM);
    float rs = rsqrtf(var + 1e-5f);
#pragma unroll
    for (int d = 0; d < DIM; ++d)
      bufX2[t * DIM + d] = (xr[d] - m) * rs * ln2_g[d] + ln2_b[d];
  }
  __syncthreads();

  // ---- 6. mlp1 + exact GELU: h = gelu(x2 @ w1^T + b1) : 4 WMMAs -----------
#pragma unroll
  for (int mt = 0; mt < 2; ++mt) {
    v16h a = load_A_k16(bufX2, DIM, mt, lrow, half);
#pragma unroll
    for (int nt = 0; nt < 2; ++nt) {
      v16h b = load_B_w(sm.wlds, 4 + nt, lane);
      v8f  c = splat8(b1[nt * 16 + lrow]);
      c = __builtin_amdgcn_wmma_f32_16x16x32_f16(false, a, false, b,
                                                 (short)0, c, false, false);
#pragma unroll
      for (int r = 0; r < 8; ++r) {
        float x = c[r];
        float g = 0.5f * x * (1.0f + erff(x * 0.70710678118654752f));
        bufQKV[(mt * 16 + half * 8 + r) * 32 + nt * 16 + lrow] = g;  // h[32][32]
      }
    }
  }
  __syncthreads();

  // ---- 7. mlp2: h @ w2^T + b2 (K=32 exact) : 2 WMMAs ----------------------
#pragma unroll
  for (int mt = 0; mt < 2; ++mt) {
    v16h a = load_A_k32(bufQKV, 32, mt, lrow, half);
    v16h b = load_B_w(sm.wlds, 6, lane);
    v8f  c = splat8(b2[lrow]);
    c = __builtin_amdgcn_wmma_f32_16x16x32_f16(false, a, false, b,
                                               (short)0, c, false, false);
#pragma unroll
    for (int r = 0; r < 8; ++r)
      bufX[(mt * 16 + half * 8 + r) * DIM + lrow] = c[r];
  }
  __syncthreads();

  // residual: x = x2 + mlp_out (lane t owns row t)
  {
    const int t = lane;
#pragma unroll
    for (int d = 0; d < DIM; ++d) bufX[t * DIM + d] += bufX2[t * DIM + d];
  }
  __syncthreads();

  // ---- 8. logits = x @ tok_emb^T : [32,16]x[16,96], 12 WMMAs --------------
  const size_t obase = (size_t)seq * SEQT * NVOC;
#pragma unroll
  for (int mt = 0; mt < 2; ++mt) {
    v16h a = load_A_k16(bufX, DIM, mt, lrow, half);
#pragma unroll
    for (int nt = 0; nt < 6; ++nt) {
      v16h b = load_B_w(sm.wlds, 7 + nt, lane);
      v8f  c = splat8(0.0f);
      c = __builtin_amdgcn_wmma_f32_16x16x32_f16(false, a, false, b,
                                                 (short)0, c, false, false);
#pragma unroll
      for (int r = 0; r < 8; ++r) {
        size_t o = obase + (size_t)(mt * 16 + half * 8 + r) * NVOC + nt * 16 + lrow;
        __builtin_nontemporal_store(c[r], out + o);   // 201MB stream: bypass L2 retention
      }
    }
  }
}

extern "C" void kernel_launch(void* const* d_in, const int* in_sizes, int n_in,
                              void* d_out, int out_size, void* d_ws, size_t ws_size,
                              hipStream_t stream) {
  const int*   idx     = (const int*)  d_in[0];
  const float* tok_emb = (const float*)d_in[1];
  const float* pos_emb = (const float*)d_in[2];
  const float* ln1_g   = (const float*)d_in[3];
  const float* ln1_b   = (const float*)d_in[4];
  const float* in_w    = (const float*)d_in[5];
  const float* in_b    = (const float*)d_in[6];
  const float* out_w   = (const float*)d_in[7];
  const float* out_b   = (const float*)d_in[8];
  const float* ln2_g   = (const float*)d_in[9];
  const float* ln2_b   = (const float*)d_in[10];
  const float* w1      = (const float*)d_in[11];
  const float* b1      = (const float*)d_in[12];
  const float* w2      = (const float*)d_in[13];
  const float* b2      = (const float*)d_in[14];
  float* out = (float*)d_out;

  const int nseq = in_sizes[0] / SEQT;                  // 16384
  const int blocks = (nseq + WPB - 1) / WPB;            // 4096
  tinygpt_fused_kernel<<<dim3(blocks), dim3(WPB * 32), 0, stream>>>(
      idx, tok_emb, pos_emb, ln1_g, ln1_b, in_w, in_b, out_w, out_b,
      ln2_g, ln2_b, w1, b1, w2, b2, out, nseq);
}
